// Conv3DFusionModule_63204738728014
// MI455X (gfx1250) — matrix operand order repaired and checked by reading, not compiled, not run on hardware
//
#include <hip/hip_runtime.h>

typedef __attribute__((ext_vector_type(16))) _Float16 v16h;
typedef __attribute__((ext_vector_type(8)))  _Float16 v8h;
typedef __attribute__((ext_vector_type(8)))  float    v8f;

union AV { v16h v; v8h h[2]; };

// ---------------------------------------------------------------- converts
__global__ void cvt_f32_f16(const float* __restrict__ in, _Float16* __restrict__ out, size_t n) {
  size_t t = (size_t)blockIdx.x * blockDim.x + threadIdx.x;
  if (t < n) out[t] = (_Float16)in[t];
}

// Pack f32 weight (taps, cin, cout) row-major into per-(k, cin/32, cout/16)
// WMMA B fragments: frag[lane][e] with lane = ((K>>4)&1)*16 | (N&15), e = K&15.
__global__ void pack_w_kernel(const float* __restrict__ w, _Float16* __restrict__ dst,
                              int taps, int cin, int cout) {
  int total = taps * cin * cout;
  int t = blockIdx.x * blockDim.x + threadIdx.x;
  if (t >= total) return;
  int per = cin * cout;
  int k  = t / per;
  int r  = t - k * per;
  int K  = r / cout;
  int Nc = r - K * cout;
  int nkc = cin >> 5, nct = cout >> 4;
  int kc = K >> 5, ct = Nc >> 4;
  int lane = (((K >> 4) & 1) << 4) | (Nc & 15);
  int e = K & 15;
  size_t d = ((((size_t)k * nkc + kc) * nct + ct) * 32 + lane) * 16 + e;
  dst[d] = (_Float16)w[t];
}

// features_at_coordinates: ya = finite_or_zero(y2[align_idx]) -> zin[:,64:96]
__global__ void align_gather_kernel(const _Float16* __restrict__ y2,
                                    const int* __restrict__ aidx,
                                    _Float16* __restrict__ zin, int n) {
  int t = blockIdx.x * blockDim.x + threadIdx.x;
  if (t >= n * 32) return;
  int i = t >> 5, c = t & 31;
  int j = aidx[i];
  float v = 0.f;
  if (j >= 0 && j < n) v = (float)y2[(size_t)j * 32 + c];
  if (!(v == v) || v > 3.0e38f || v < -3.0e38f) v = 0.f;
  zin[(size_t)i * 96 + 64 + c] = (_Float16)v;
}

// ---------------------------------------------------------------- main layer
// Block = 8 waves x 32 rows = 256 voxel rows. Per tap: all NKC A fragments for
// both 16-row tiles are gathered up-front (one masked clause each), and the
// NKC*NCT B fragments stream from LDS through a parity-indexed 2-deep register
// pipeline (no rotate copies -> no WMMA WAR-hazard NOPs); each fragment's ds
// latency hides under the previous WMMA pair. B staging is LDS double-buffered
// across taps with the copy overlapped with compute; neighbor indices are
// software-pipelined one tap ahead. Tail handled with data guards only
// (barriers stay uniform). Epilogue: relu(x*scale + shift), f16/f32 out.
template<int CIN, int COUT, int TAPS, bool OUTF32>
__global__ __launch_bounds__(256) void layer_wmma(
    const _Float16* __restrict__ xin,
    const _Float16* __restrict__ wpk,
    const float* __restrict__ bns,
    const float* __restrict__ bnb,
    const int* __restrict__ nbr,
    void* __restrict__ outp,
    int n, int ostride) {
  constexpr int NKC = CIN / 32;        // 32-deep K chunks
  constexpr int NCT = COUT / 16;       // 16-wide N tiles
  constexpr int NFR = NKC * NCT;       // B fragments per tap
  constexpr int STG = NFR * 512;       // f16 elems per staged tap
  constexpr int NBUF = (TAPS > 1) ? 2 : 1;
  __shared__ _Float16 sB[NBUF * STG];

  const int lane = threadIdx.x & 31;
  const int wave = threadIdx.x >> 5;
  const int row0 = (blockIdx.x * 8 + wave) * 32;
  const int m  = lane & 15;            // matrix row / output column within tile
  const int hh = lane >> 4;            // lane-half selector
  const int vA = row0 + m;             // row tile 0
  const int vB = row0 + 16 + m;        // row tile 1
  const bool inA = vA < n;             // tail guards (data-level only)
  const bool inB = vB < n;

  // cooperative global->LDS stage of one tap's packed B fragments
  auto stage = [&](int kk) {
    const uint4* src = (const uint4*)(wpk + (size_t)kk * STG);
    uint4* dst = (uint4*)(sB + (kk & (NBUF - 1)) * STG);
    for (int i = threadIdx.x; i < STG / 8; i += 256) dst[i] = src[i];
  };

  v8f zf = {};
  v8f acc0[NCT], acc1[NCT];
#pragma unroll
  for (int ct = 0; ct < NCT; ++ct) { acc0[ct] = zf; acc1[ct] = zf; }

  stage(0);
  int idxA = -1, idxB = -1;
  if (TAPS > 1) {
    if (inA) idxA = nbr[vA];
    if (inB) idxB = nbr[vB];
  } else {
    idxA = inA ? vA : -1;
    idxB = inB ? vB : -1;
  }
  __syncthreads();

#pragma unroll 1
  for (int k = 0; k < TAPS; ++k) {
    if (TAPS > 1 && k + 1 < TAPS) stage(k + 1);          // overlap copy w/ compute

    const int curA = idxA, curB = idxB;
    if (TAPS > 1 && k + 1 < TAPS) {                      // pipeline next indices
      idxA = inA ? nbr[(size_t)(k + 1) * n + vA] : -1;
      idxB = inB ? nbr[(size_t)(k + 1) * n + vB] : -1;
    }
    const bool vaA = curA >= 0, vaB = curB >= 0;
    const _Float16* rpA = xin + (size_t)(vaA ? curA : 0) * CIN;
    const _Float16* rpB = xin + (size_t)(vaB ? curB : 0) * CIN;
    const _Float16* sbuf = sB + (k & (NBUF - 1)) * STG;

    // hoist ALL A fragments for this tap (one masked load clause per tile)
    // A (16x32 f16): lanes 0-15 hold K 0-15, lanes 16-31: 16-31
    AV a0f[NKC], a1f[NKC];
    v8h zh = {};
#pragma unroll
    for (int kc = 0; kc < NKC; ++kc) {
      a0f[kc].h[0] = zh; a0f[kc].h[1] = zh;
      a1f[kc].h[0] = zh; a1f[kc].h[1] = zh;
    }
    if (vaA) {
#pragma unroll
      for (int kc = 0; kc < NKC; ++kc) {
        const _Float16* s = rpA + kc * 32 + hh * 16;
        a0f[kc].h[0] = *(const v8h*)(s);
        a0f[kc].h[1] = *(const v8h*)(s + 8);
      }
    }
    if (vaB) {
#pragma unroll
      for (int kc = 0; kc < NKC; ++kc) {
        const _Float16* s = rpB + kc * 32 + hh * 16;
        a1f[kc].h[0] = *(const v8h*)(s);
        a1f[kc].h[1] = *(const v8h*)(s + 8);
      }
    }

    // B register pipeline: parity ping-pong, load fragment t+1 before using t
    auto ldb = [&](AV& b, int t) {
      const v8h* bp = (const v8h*)(sbuf + ((size_t)t * 32 + lane) * 16);
      b.h[0] = bp[0];
      b.h[1] = bp[1];
    };
    AV bf[2];
    ldb(bf[0], 0);
#pragma unroll
    for (int t = 0; t < NFR; ++t) {
      if (t + 1 < NFR) ldb(bf[(t + 1) & 1], t + 1);
      const int kc = t / NCT, ct = t % NCT;
      acc0[ct] = __builtin_amdgcn_wmma_f32_16x16x32_f16(
          false, a0f[kc].v, false, bf[t & 1].v, (short)0, acc0[ct], false, false);
      acc1[ct] = __builtin_amdgcn_wmma_f32_16x16x32_f16(
          false, a1f[kc].v, false, bf[t & 1].v, (short)0, acc1[ct], false, false);
    }

    if (TAPS > 1) __syncthreads();  // next-tap copy done & current buffer released
  }

  // C/D layout: VGPR r -> row = base + r + hh*8, col = ct*16 + m
#pragma unroll
  for (int ct = 0; ct < NCT; ++ct) {
    const int nc = ct * 16 + m;
    const float sc = bns[nc];
    const float sb = bnb[nc];
#pragma unroll
    for (int r = 0; r < 8; ++r) {
      float v0 = acc0[ct][r] * sc + sb;
      float v1 = acc1[ct][r] * sc + sb;
      v0 = v0 > 0.f ? v0 : 0.f;
      v1 = v1 > 0.f ? v1 : 0.f;
      const int rowA = row0 + r + hh * 8;
      const int rowB = rowA + 16;
      if (rowA < n) {
        const size_t o0 = (size_t)rowA * ostride + nc;
        if (OUTF32) ((float*)outp)[o0] = v0; else ((_Float16*)outp)[o0] = (_Float16)v0;
      }
      if (rowB < n) {
        const size_t o1 = (size_t)rowB * ostride + nc;
        if (OUTF32) ((float*)outp)[o1] = v1; else ((_Float16*)outp)[o1] = (_Float16)v1;
      }
    }
  }
}

// ---------------------------------------------------------------- launcher
extern "C" void kernel_launch(void* const* d_in, const int* in_sizes, int n_in,
                              void* d_out, int out_size, void* d_ws, size_t ws_size,
                              hipStream_t stream) {
  (void)in_sizes; (void)n_in; (void)out_size; (void)ws_size;
  const int N = 80000;

  const float* feat3d = (const float*)d_in[0];
  const float* feat2d = (const float*)d_in[1];
  const float* W[11];
  for (int i = 0; i < 11; ++i) W[i] = (const float*)d_in[2 + i];
  const float *BNs[11], *BNb[11];
  for (int i = 0; i < 11; ++i) { BNs[i] = (const float*)d_in[13 + 2*i]; BNb[i] = (const float*)d_in[14 + 2*i]; }
  const int* nbr  = (const int*)d_in[35];
  const int* aidx = (const int*)d_in[36];

  // Workspace layout (bytes). zin doubles as h3d (both N x 96 f16); h2d region
  // is recycled for the 128-wide ping buffer after b1 consumes feat2d.
  char* ws = (char*)d_ws;
  _Float16* wpk   = (_Float16*)(ws);                              // 2.3 MB packed f16 weights
  _Float16* zin   = (_Float16*)(ws + (size_t)4  * 1024 * 1024);   // h3d, then concat [x3|ya]
  _Float16* h2d   = (_Float16*)(ws + (size_t)20 * 1024 * 1024);   // feat2d f16 (41 MB)
  _Float16* t128a = (_Float16*)(ws + (size_t)20 * 1024 * 1024);   // reuse after b1
  _Float16* t128b = (_Float16*)(ws + (size_t)40 * 1024 * 1024);
  _Float16* tA    = (_Float16*)(ws + (size_t)64 * 1024 * 1024);   // N x 64 f16
  _Float16* tB    = (_Float16*)(ws + (size_t)76 * 1024 * 1024);   // N x 64 f16

  static const int taps [11] = {27, 1,27, 1,27, 1,27, 1,27,  1,  1};
  static const int cins [11] = {96,64,64,64,256,64,64,32,96,128,128};
  static const int couts[11] = {64,64,64,64, 64,64,32,32,128,128,128};
  size_t woff[12]; woff[0] = 0;
  for (int i = 0; i < 11; ++i) woff[i+1] = woff[i] + (size_t)taps[i]*cins[i]*couts[i];

  // f32 -> f16 input conversion
  {
    size_t n3 = (size_t)N * 96;
    cvt_f32_f16<<<(unsigned)((n3 + 255) / 256), 256, 0, stream>>>(feat3d, zin, n3);
    size_t n2 = (size_t)N * 256;
    cvt_f32_f16<<<(unsigned)((n2 + 255) / 256), 256, 0, stream>>>(feat2d, h2d, n2);
  }
  // weight packing
  for (int i = 0; i < 11; ++i) {
    int tot = taps[i] * cins[i] * couts[i];
    pack_w_kernel<<<(tot + 255) / 256, 256, 0, stream>>>(W[i], wpk + woff[i],
                                                         taps[i], cins[i], couts[i]);
  }

  dim3 blk(256);             // 8 waves x 32 rows = 256 rows per block
  dim3 grd((N + 255) / 256); // 313 blocks; tail via data guards (no early return)

  // 3D branch: 96 -> 64
  layer_wmma< 96, 64,27,false><<<grd,blk,0,stream>>>(zin,   wpk+woff[0], BNs[0], BNb[0], nbr, tA,   N, 64);
  layer_wmma< 64, 64, 1,false><<<grd,blk,0,stream>>>(tA,    wpk+woff[1], BNs[1], BNb[1], nbr, tB,   N, 64);
  layer_wmma< 64, 64,27,false><<<grd,blk,0,stream>>>(tB,    wpk+woff[2], BNs[2], BNb[2], nbr, tA,   N, 64);
  layer_wmma< 64, 64, 1,false><<<grd,blk,0,stream>>>(tA,    wpk+woff[3], BNs[3], BNb[3], nbr, zin,  N, 96); // x3 -> zin[:,0:64]
  // 2D branch: 256 -> 32
  layer_wmma<256, 64,27,false><<<grd,blk,0,stream>>>(h2d,   wpk+woff[4], BNs[4], BNb[4], nbr, tA,   N, 64);
  layer_wmma< 64, 64, 1,false><<<grd,blk,0,stream>>>(tA,    wpk+woff[5], BNs[5], BNb[5], nbr, tB,   N, 64);
  layer_wmma< 64, 32,27,false><<<grd,blk,0,stream>>>(tB,    wpk+woff[6], BNs[6], BNb[6], nbr, tA,   N, 32);
  layer_wmma< 32, 32, 1,false><<<grd,blk,0,stream>>>(tA,    wpk+woff[7], BNs[7], BNb[7], nbr, tB,   N, 32); // y2
  align_gather_kernel<<<(N * 32 + 255) / 256, 256, 0, stream>>>(tB, aidx, zin, N);       // ya -> zin[:,64:96]
  // fusion: 96 -> 128
  layer_wmma< 96,128,27,false><<<grd,blk,0,stream>>>(zin,   wpk+woff[8],  BNs[8],  BNb[8],  nbr, t128a, N, 128);
  layer_wmma<128,128, 1,false><<<grd,blk,0,stream>>>(t128a, wpk+woff[9],  BNs[9],  BNb[9],  nbr, t128b, N, 128);
  layer_wmma<128,128, 1,true ><<<grd,blk,0,stream>>>(t128b, wpk+woff[10], BNs[10], BNb[10], nbr, d_out, N, 128);
}
